// MyCodebookEMA_65575560675469
// MI455X (gfx1250) — compile-verified, matrix operand order checked by reading, not run on hardware
//
#include <hip/hip_runtime.h>
#include <hip/hip_bf16.h>

// ---------------------------------------------------------------------------
// VQ-VAE codebook quantization + EMA update for MI455X (gfx1250, wave32).
// Distance GEMM [16384,256]x[256,1024] runs on v_wmma_f32_16x16x32_bf16.
// M-tile = 32 rows/block: each B fragment feeds two interleaved WMMA chains.
// ---------------------------------------------------------------------------

#define N_CODES   1024
#define EMB_DIM   256
#define BHW       16384          // 16 * 32 * 32
#define DECAY     0.99f
#define ONE_M_D   0.01f
#define EPSV      1e-5f

typedef __attribute__((ext_vector_type(16))) __bf16 v16bf;
typedef __attribute__((ext_vector_type(8)))  float  v8f;

union FragU { uint4 u[2]; v16bf v; };

__device__ __forceinline__ unsigned short f2bf(float x) {
    unsigned int u = __float_as_uint(x);
    unsigned int r = u + 0x7fffu + ((u >> 16) & 1u);   // round-to-nearest-even
    return (unsigned short)(r >> 16);
}

// ---------------------------------------------------------------------------
// Zero scratch accumulators (counts + dw + loss, contiguous).
// ---------------------------------------------------------------------------
__global__ void vq_zero(float* p, int n) {
    int i = blockIdx.x * 256 + threadIdx.x;
    if (i < n) p[i] = 0.0f;
}

// ---------------------------------------------------------------------------
// z [16,256,32,32] f32 -> zp_bf [16384,256] bf16 (transpose) + z_sq [16384].
// One block = 16 consecutive rows (same b,h; w = 0..15 slice).
// ---------------------------------------------------------------------------
__global__ void vq_prep_z(const float* __restrict__ z,
                          unsigned short* __restrict__ zp_bf,
                          float* __restrict__ z_sq) {
    const int m0  = blockIdx.x * 16;
    const int b   = m0 >> 10;          // / 1024
    const int hw0 = m0 & 1023;
    const int t   = threadIdx.x;

    __shared__ float tile[16][EMB_DIM + 1];

    const int w   = t & 15;
    const int chi = t >> 4;
    #pragma unroll 4
    for (int it = 0; it < 16; ++it) {
        int c = it * 16 + chi;
        tile[w][c] = z[(size_t)b * (EMB_DIM * 1024) + (size_t)c * 1024 + hw0 + w];
    }
    __syncthreads();

    const int r  = t >> 4;             // row in tile
    const int c0 = (t & 15) * 16;
    float ss = 0.0f;
    #pragma unroll
    for (int j = 0; j < 16; ++j) {
        float v = tile[r][c0 + j];
        ss += v * v;
        zp_bf[(size_t)(m0 + r) * EMB_DIM + c0 + j] = f2bf(v);
    }
    #pragma unroll
    for (int off = 1; off < 16; off <<= 1) ss += __shfl_xor(ss, off, 32);
    if ((t & 15) == 0) z_sq[m0 + r] = ss;
}

// ---------------------------------------------------------------------------
// codebook [1024,256] f32 -> bf16 + c_sq [1024]. One block per code row.
// ---------------------------------------------------------------------------
__global__ void vq_prep_cb(const float* __restrict__ cb,
                           unsigned short* __restrict__ cb_bf,
                           float* __restrict__ c_sq) {
    const int n = blockIdx.x;
    const int t = threadIdx.x;                 // 256 threads
    float v = cb[(size_t)n * EMB_DIM + t];
    cb_bf[(size_t)n * EMB_DIM + t] = f2bf(v);
    float ss = v * v;
    #pragma unroll
    for (int off = 1; off < 32; off <<= 1) ss += __shfl_xor(ss, off, 32);
    __shared__ float wsum[8];
    if ((t & 31) == 0) wsum[t >> 5] = ss;
    __syncthreads();
    if (t == 0) {
        float s = 0.0f;
        #pragma unroll
        for (int i = 0; i < 8; ++i) s += wsum[i];
        c_sq[n] = s;
    }
}

// ---------------------------------------------------------------------------
// Main: distances via WMMA, argmin, idx/zq/loss/histogram/dw-scatter.
// Block = 256 threads (8 waves), handles 32 z-rows vs all 1024 codes.
// Each wave: 2 resident A tiles (rows m0..m0+15, m0+16..m0+31), sweeps
// 8 N-tiles; B fragments preloaded per tile, then 2 interleaved WMMA chains.
// ---------------------------------------------------------------------------
__global__ void __launch_bounds__(256)
vq_main(const unsigned short* __restrict__ zp_bf,
        const unsigned short* __restrict__ cb_bf,
        const float* __restrict__ z_sq,
        const float* __restrict__ c_sq,
        const float* __restrict__ codebook,
        const float* __restrict__ z,
        float* __restrict__ o_zq,
        float* __restrict__ o_idx,
        float* __restrict__ counts,
        float* __restrict__ dw,
        float* __restrict__ loss_acc) {
    const int m0    = blockIdx.x * 32;
    const int t     = threadIdx.x;
    const int wave  = t >> 5;
    const int lane  = t & 31;
    const int lrow  = lane & 15;               // M index (A) / N index (B)
    const int khalf = (lane < 16) ? 0 : 8;     // ISA 16-bit fragment K split

    // ---- two resident A tiles: 32 rows x K=256, 8 chunks of K=32 each -----
    v16bf afr0[8], afr1[8];
    {
        const unsigned short* ar0 = zp_bf + (size_t)(m0 + lrow) * EMB_DIM;
        const unsigned short* ar1 = ar0 + 16 * EMB_DIM;
        #pragma unroll
        for (int kc = 0; kc < 8; ++kc) {
            FragU f0, f1;
            const unsigned short* p0 = ar0 + kc * 32 + khalf;
            const unsigned short* p1 = ar1 + kc * 32 + khalf;
            f0.u[0] = *(const uint4*)(p0);        // K +0..7   -> v0..v3
            f0.u[1] = *(const uint4*)(p0 + 16);   // K +16..23 -> v4..v7
            f1.u[0] = *(const uint4*)(p1);
            f1.u[1] = *(const uint4*)(p1 + 16);
            afr0[kc] = f0.v;
            afr1[kc] = f1.v;
        }
    }

    float best_d0[8], best_d1[8];
    int   best_n0[8], best_n1[8];
    #pragma unroll
    for (int v = 0; v < 8; ++v) {
        best_d0[v] = 3.4e38f; best_n0[v] = 0;
        best_d1[v] = 3.4e38f; best_n1[v] = 0;
    }

    // ---- sweep all 1024 codes: wave w owns tiles n = iter*128 + w*16 ------
    for (int iter = 0; iter < 8; ++iter) {
        const int n = iter * 128 + wave * 16 + lrow;
        const unsigned short* brow = cb_bf + (size_t)n * EMB_DIM;

        // preload all B fragments for this tile (lets loads clause + overlap)
        FragU bf[8];
        #pragma unroll
        for (int kc = 0; kc < 8; ++kc) {
            const unsigned short* p = brow + kc * 32 + khalf;
            bf[kc].u[0] = *(const uint4*)(p);
            bf[kc].u[1] = *(const uint4*)(p + 16);
        }

        v8f acc0 = {}, acc1 = {};
        #pragma unroll
        for (int kc = 0; kc < 8; ++kc) {
            acc0 = __builtin_amdgcn_wmma_f32_16x16x32_bf16(
                false, afr0[kc], false, bf[kc].v, (short)0, acc0, false, false);
            acc1 = __builtin_amdgcn_wmma_f32_16x16x32_bf16(
                false, afr1[kc], false, bf[kc].v, (short)0, acc1, false, false);
        }

        const float csn = c_sq[n];
        #pragma unroll
        for (int v = 0; v < 8; ++v) {
            float d0 = csn - 2.0f * acc0[v];   // z_sq is row-constant, skip it
            float d1 = csn - 2.0f * acc1[v];
            if (d0 < best_d0[v] || (d0 == best_d0[v] && n < best_n0[v])) {
                best_d0[v] = d0; best_n0[v] = n;
            }
            if (d1 < best_d1[v] || (d1 == best_d1[v] && n < best_n1[v])) {
                best_d1[v] = d1; best_n1[v] = n;
            }
        }
    }

    // ---- argmin reduction across the 16 lanes holding this tile's N -------
    #pragma unroll
    for (int off = 1; off < 16; off <<= 1) {
        #pragma unroll
        for (int v = 0; v < 8; ++v) {
            float od = __shfl_xor(best_d0[v], off, 32);
            int   on = __shfl_xor(best_n0[v], off, 32);
            if (od < best_d0[v] || (od == best_d0[v] && on < best_n0[v])) {
                best_d0[v] = od; best_n0[v] = on;
            }
            od = __shfl_xor(best_d1[v], off, 32);
            on = __shfl_xor(best_n1[v], off, 32);
            if (od < best_d1[v] || (od == best_d1[v] && on < best_n1[v])) {
                best_d1[v] = od; best_n1[v] = on;
            }
        }
    }

    __shared__ float s_d[8][32];
    __shared__ int   s_n[8][32];
    __shared__ int   fin_n[32];
    if (lrow == 0) {                           // lanes 0 and 16 of each wave
        const int rbase = (lane < 16) ? 0 : 8; // C-layout: lanes16-31 hold M=v+8
        #pragma unroll
        for (int v = 0; v < 8; ++v) {
            s_d[wave][rbase + v]      = best_d0[v];
            s_n[wave][rbase + v]      = best_n0[v];
            s_d[wave][16 + rbase + v] = best_d1[v];
            s_n[wave][16 + rbase + v] = best_n1[v];
        }
    }
    __syncthreads();

    // ---- cross-wave reduction, final per-row results -----------------------
    if (t < 32) {
        float bd = s_d[0][t];
        int   bn = s_n[0][t];
        #pragma unroll
        for (int w2 = 1; w2 < 8; ++w2) {
            float dd = s_d[w2][t];
            int   nn = s_n[w2][t];
            if (dd < bd || (dd == bd && nn < bn)) { bd = dd; bn = nn; }
        }
        fin_n[t] = bn;
        o_idx[m0 + t] = (float)bn;
        atomicAdd(&counts[bn], 1.0f);
        float lpart = z_sq[m0 + t] + bd;       // ||z - q||^2 for this row
        #pragma unroll
        for (int off = 1; off < 32; off <<= 1) lpart += __shfl_xor(lpart, off, 32);
        if (t == 0) atomicAdd(loss_acc, lpart);
    }
    __syncthreads();

    // ---- write zq (NCHW, == straight-through value) + dw scatter ----------
    const int b   = m0 >> 10;
    const int hw0 = m0 & 1023;
    const int wl  = t & 31;                    // row within tile == w offset
    const int cg  = t >> 5;                    // 0..7
    const int nq  = fin_n[wl];
    #pragma unroll 2
    for (int it = 0; it < 32; ++it) {
        int c = it * 8 + cg;
        size_t zidx = (size_t)b * (EMB_DIM * 1024) + (size_t)c * 1024 + hw0 + wl;
        o_zq[zidx] = codebook[(size_t)nq * EMB_DIM + c];
        atomicAdd(&dw[(size_t)nq * EMB_DIM + c], z[zidx]);
    }
}

// ---------------------------------------------------------------------------
// Laplace-smoothed cluster sizes + final loss scale. One block, 1024 threads.
// ---------------------------------------------------------------------------
__global__ void vq_finalize_cs(const float* __restrict__ counts,
                               const float* __restrict__ ema_cs,
                               const float* __restrict__ loss_acc,
                               float* __restrict__ o_loss,
                               float* __restrict__ o_cs,
                               float* __restrict__ ws_cs) {
    const int t = threadIdx.x;                 // 1024 threads = 32 waves
    float cs = ema_cs[t] * DECAY + ONE_M_D * counts[t];
    float s = cs;
    #pragma unroll
    for (int off = 1; off < 32; off <<= 1) s += __shfl_xor(s, off, 32);
    __shared__ float sbuf[32];
    __shared__ float n_tot;
    if ((t & 31) == 0) sbuf[t >> 5] = s;
    __syncthreads();
    if (t == 0) {
        float tt = 0.0f;
        #pragma unroll
        for (int i = 0; i < 32; ++i) tt += sbuf[i];
        n_tot = tt;
        o_loss[0] = 0.25f * loss_acc[0] / ((float)BHW * (float)EMB_DIM);
    }
    __syncthreads();
    float nt = n_tot;
    float cs_s = (cs + EPSV) / (nt + (float)N_CODES * EPSV) * nt;
    o_cs[t]  = cs_s;
    ws_cs[t] = cs_s;
}

// ---------------------------------------------------------------------------
// EMA weight update + new codebook. Elementwise over 1024*256.
// ---------------------------------------------------------------------------
__global__ void vq_finalize_w(const float* __restrict__ ema_w,
                              const float* __restrict__ dw,
                              const float* __restrict__ ws_cs,
                              float* __restrict__ o_new_cb,
                              float* __restrict__ o_new_ema_w) {
    int i = blockIdx.x * 256 + threadIdx.x;    // 262144 elements
    float nw = ema_w[i] * DECAY + ONE_M_D * dw[i];
    o_new_ema_w[i] = nw;
    o_new_cb[i]    = nw / ws_cs[i >> 8];
}

// ---------------------------------------------------------------------------
extern "C" void kernel_launch(void* const* d_in, const int* in_sizes, int n_in,
                              void* d_out, int out_size, void* d_ws, size_t ws_size,
                              hipStream_t stream) {
    (void)in_sizes; (void)n_in; (void)out_size; (void)ws_size;

    const float* z        = (const float*)d_in[0];   // [16,256,32,32]
    const float* codebook = (const float*)d_in[1];   // [1024,256]
    const float* ema_cs   = (const float*)d_in[2];   // [1024]
    const float* ema_w    = (const float*)d_in[3];   // [1024,256]

    // outputs, concatenated flat in return order (all as float)
    float* out      = (float*)d_out;
    float* o_loss   = out;                                   // 1
    float* o_zq     = o_loss + 1;                            // 16*256*32*32
    float* o_idx    = o_zq + (size_t)BHW * EMB_DIM;          // 16384
    float* o_ncb    = o_idx + BHW;                           // 1024*256
    float* o_cs     = o_ncb + (size_t)N_CODES * EMB_DIM;     // 1024
    float* o_new_w  = o_cs + N_CODES;                        // 1024*256

    // scratch layout
    char* ws = (char*)d_ws;
    size_t off = 0;
    unsigned short* zp_bf = (unsigned short*)(ws + off); off += (size_t)BHW * EMB_DIM * 2;
    unsigned short* cb_bf = (unsigned short*)(ws + off); off += (size_t)N_CODES * EMB_DIM * 2;
    float* z_sq   = (float*)(ws + off); off += (size_t)BHW * 4;
    float* c_sq   = (float*)(ws + off); off += (size_t)N_CODES * 4;
    float* counts = (float*)(ws + off); off += (size_t)N_CODES * 4;       // zeroed
    float* dw     = (float*)(ws + off); off += (size_t)N_CODES * EMB_DIM * 4; // zeroed
    float* loss_a = (float*)(ws + off); off += 4;                         // zeroed
    float* ws_cs  = (float*)(ws + off); off += (size_t)N_CODES * 4;

    const int n_zero = N_CODES + N_CODES * EMB_DIM + 1;      // counts..loss contiguous
    vq_zero<<<(n_zero + 255) / 256, 256, 0, stream>>>(counts, n_zero);

    vq_prep_z<<<BHW / 16, 256, 0, stream>>>(z, zp_bf, z_sq);
    vq_prep_cb<<<N_CODES, 256, 0, stream>>>(codebook, cb_bf, c_sq);

    vq_main<<<BHW / 32, 256, 0, stream>>>(zp_bf, cb_bf, z_sq, c_sq, codebook, z,
                                          o_zq, o_idx, counts, dw, loss_a);

    vq_finalize_cs<<<1, 1024, 0, stream>>>(counts, ema_cs, loss_a, o_loss, o_cs, ws_cs);
    vq_finalize_w<<<(N_CODES * EMB_DIM) / 256, 256, 0, stream>>>(ema_w, dw, ws_cs,
                                                                 o_ncb, o_new_w);
}
